// GilmerNet_10926396801337
// MI455X (gfx1250) — compile-verified, matrix operand order checked by reading, not run on hardware
//
#include <hip/hip_runtime.h>
#include <hip/hip_bf16.h>
#include <math.h>

// ---------------- problem constants (match reference) ----------------
#define NN      16384   // nodes
#define EE      49152   // edges
#define NG      512     // graphs
#define DIM     64
#define NF      11
#define EF      4
#define HID     128
#define EWN     4096    // DIM*DIM
#define JUMPS   3
#define S2S     3

typedef __attribute__((ext_vector_type(16))) __bf16 v16bf;
typedef __attribute__((ext_vector_type(8)))  float  v8f;

// -------- bf16 <-> f32 helpers (bit-level, RNE) --------
__device__ __forceinline__ unsigned short f2bf(float f) {
    unsigned u = __float_as_uint(f);
    u += 0x7FFFu + ((u >> 16) & 1u);
    return (unsigned short)(u >> 16);
}
__device__ __forceinline__ float bf2f(unsigned short s) {
    return __uint_as_float(((unsigned)s) << 16);
}
__device__ __forceinline__ float sigm(float x) { return 1.0f / (1.0f + __expf(-x)); }

// ---------------- generic zero fill ----------------
__global__ void k_zero(float* __restrict__ p, int n) {
    int i = blockIdx.x * blockDim.x + threadIdx.x;
    if (i < n) p[i] = 0.0f;
}

// ---------------- convert mlp_w2 (f32 [4096,128]) -> bf16 (same layout, K contiguous) ------
__global__ void k_w2bf(const float* __restrict__ w2, unsigned short* __restrict__ w2bf, int n) {
    int i = blockIdx.x * blockDim.x + threadIdx.x;
    if (i < n) w2bf[i] = f2bf(w2[i]);
}

// ---------------- lin0: out = relu(x @ lin0_w.T + b); h = out ----------------
__global__ void k_lin0(const float* __restrict__ x, const float* __restrict__ w,
                       const float* __restrict__ b, float* __restrict__ out,
                       float* __restrict__ h) {
    int idx = blockIdx.x * blockDim.x + threadIdx.x;       // node*64 + o
    if (idx >= NN * DIM) return;
    int n = idx >> 6, o = idx & 63;
    float s = b[o];
    const float* xr = x + (size_t)n * NF;
    const float* wr = w + (size_t)o * NF;
    #pragma unroll
    for (int f = 0; f < NF; ++f) s += xr[f] * wr[f];
    s = fmaxf(s, 0.0f);
    out[idx] = s;
    h[idx]   = s;
}

// ---------------- edge MLP layer 1: hid = relu(ea @ w1.T + b1), stored bf16 -----------
__global__ void k_edgehid(const float* __restrict__ ea, const float* __restrict__ w1,
                          const float* __restrict__ b1, unsigned short* __restrict__ hidbf) {
    int idx = blockIdx.x * blockDim.x + threadIdx.x;       // e*128 + j
    if (idx >= EE * HID) return;
    int e = idx >> 7, j = idx & 127;
    const float* er = ea + (size_t)e * EF;
    const float* wr = w1 + (size_t)j * EF;
    float s = b1[j];
    #pragma unroll
    for (int f = 0; f < EF; ++f) s += er[f] * wr[f];
    hidbf[idx] = f2bf(fmaxf(s, 0.0f));
}

// ---------------- edge MLP layer 2 (WMMA bf16): ew[E,4096] = hid @ w2.T + b2 ---------
// One wave handles TWO 16-edge M-tiles (B fragment reused twice -> 8 WMMAs per N-tile,
// independent WMMAs overlap the next chunk's loads). K=128 as four K32 chunks, A in regs.
union FragBF { uint4 q[2]; v16bf v; };

__global__ __launch_bounds__(32)
void k_ew_wmma(const unsigned short* __restrict__ hidbf,
               const unsigned short* __restrict__ w2bf,
               const float* __restrict__ b2,
               unsigned short* __restrict__ ewbf) {
    const int lane = threadIdx.x;            // 0..31
    const int m0   = blockIdx.x * 32;        // base of two 16-edge tiles
    const int half = lane >> 4;              // 0 | 1
    const int loK  = half * 8;               // K sub-offset per ISA 16-bit A layout
    const int rc   = lane & 15;              // row (A) / col (B,D)

    // A fragments for both M-tiles, all 4 K-chunks, held in registers
    FragBF a0[4], a1[4];
    const unsigned short* ap0 = hidbf + (size_t)(m0 + rc) * HID + loK;
    const unsigned short* ap1 = hidbf + (size_t)(m0 + 16 + rc) * HID + loK;
    #pragma unroll
    for (int kc = 0; kc < 4; ++kc) {
        a0[kc].q[0] = *(const uint4*)(ap0 + kc * 32);        // K = k0+lo .. +7
        a0[kc].q[1] = *(const uint4*)(ap0 + kc * 32 + 16);   // K = k0+16+lo .. +7
        a1[kc].q[0] = *(const uint4*)(ap1 + kc * 32);
        a1[kc].q[1] = *(const uint4*)(ap1 + kc * 32 + 16);
    }

    for (int nt = 0; nt < 32; ++nt) {
        const int n = (blockIdx.y * 32 + nt) * 16 + rc;      // output channel (0..4095)
        const unsigned short* bp = w2bf + (size_t)n * HID + loK;

        // fetch all 4 B chunks up front (one clause; waits can stage)
        FragBF b[4];
        #pragma unroll
        for (int kc = 0; kc < 4; ++kc) {
            b[kc].q[0] = *(const uint4*)(bp + kc * 32);
            b[kc].q[1] = *(const uint4*)(bp + kc * 32 + 16);
        }
        if (nt < 31) __builtin_prefetch(bp + (size_t)16 * HID, 0, 1);  // next tile's B row

        const float bias = b2[n];
        v8f c0, c1;
        #pragma unroll
        for (int i = 0; i < 8; ++i) { c0[i] = bias; c1[i] = bias; }

        #pragma unroll
        for (int kc = 0; kc < 4; ++kc) {
            c0 = __builtin_amdgcn_wmma_f32_16x16x32_bf16(
                     false, a0[kc].v, false, b[kc].v, (short)0, c0, false, false);
            c1 = __builtin_amdgcn_wmma_f32_16x16x32_bf16(
                     false, a1[kc].v, false, b[kc].v, (short)0, c1, false, false);
        }
        // D: lane column n, rows = v + 8*half (tile0), +16 (tile1)
        unsigned short* dp0 = ewbf + (size_t)(m0 + half * 8) * EWN + n;
        unsigned short* dp1 = ewbf + (size_t)(m0 + 16 + half * 8) * EWN + n;
        #pragma unroll
        for (int r = 0; r < 8; ++r) {
            dp0[(size_t)r * EWN] = f2bf(c0[r]);
            dp1[(size_t)r * EWN] = f2bf(c1[r]);
        }
    }
}

// ---------------- degree: deg[dst] += 1 ----------------
__global__ void k_deg(const int* __restrict__ ei, float* __restrict__ deg) {
    int e = blockIdx.x * blockDim.x + threadIdx.x;
    if (e >= EE) return;
    atomicAdd(&deg[ei[EE + e]], 1.0f);
}

// ---------------- per-edge matvec + scatter: agg[dst] += out[src] @ ew[e] -----------
// One wave per edge. Lane l owns channels 8*(l&7)..+7 for rows i = 4k + (l>>3):
// b128 loads (512B coalesced per wave-iteration), bf16 unpack via shift/mask,
// 2-stage shfl_xor cross-lane reduce, then 8 lanes issue the 64 f32 atomics.
__global__ __launch_bounds__(32)
void k_msg(const float* __restrict__ out, const unsigned short* __restrict__ ewbf,
           const int* __restrict__ ei, float* __restrict__ agg) {
    __shared__ float s_src[DIM];
    const int e = blockIdx.x;
    const int l = threadIdx.x;               // 0..31
    const int src = ei[e];
    const int dst = ei[EE + e];
    const float2 sv = *(const float2*)(out + (size_t)src * DIM + 2 * l);
    s_src[2 * l]     = sv.x;
    s_src[2 * l + 1] = sv.y;
    __syncthreads();

    const int chunk = l & 7;                 // channel group (8 channels, 16B)
    const int isub  = l >> 3;                // input-row subgroup 0..3
    float acc[8];
    #pragma unroll
    for (int j = 0; j < 8; ++j) acc[j] = 0.0f;

    const unsigned short* base = ewbf + (size_t)e * EWN + chunk * 8;
    #pragma unroll 2
    for (int it = 0; it < 16; ++it) {
        const int i = it * 4 + isub;
        const uint4 w = *(const uint4*)(base + (size_t)i * DIM);
        const float si = s_src[i];
        acc[0] += si * __uint_as_float(w.x << 16);
        acc[1] += si * __uint_as_float(w.x & 0xffff0000u);
        acc[2] += si * __uint_as_float(w.y << 16);
        acc[3] += si * __uint_as_float(w.y & 0xffff0000u);
        acc[4] += si * __uint_as_float(w.z << 16);
        acc[5] += si * __uint_as_float(w.z & 0xffff0000u);
        acc[6] += si * __uint_as_float(w.w << 16);
        acc[7] += si * __uint_as_float(w.w & 0xffff0000u);
    }
    #pragma unroll
    for (int j = 0; j < 8; ++j) {
        acc[j] += __shfl_xor(acc[j], 8, 32);
        acc[j] += __shfl_xor(acc[j], 16, 32);
    }
    if (l < 8) {
        float* ap = agg + (size_t)dst * DIM + 8 * l;
        #pragma unroll
        for (int j = 0; j < 8; ++j) atomicAdd(&ap[j], acc[j]);
    }
}

// ---------------- node update: m = relu(agg/deg + out@root + bias); h = GRU(m,h) ----
__global__ __launch_bounds__(64)
void k_node(const float* __restrict__ agg, const float* __restrict__ deg,
            const float* __restrict__ root, const float* __restrict__ cbias,
            const float* __restrict__ w_ih, const float* __restrict__ w_hh,
            const float* __restrict__ b_ih, const float* __restrict__ b_hh,
            float* __restrict__ out, float* __restrict__ h) {
    __shared__ float s_out[DIM], s_h[DIM], s_m[DIM];
    const int n = blockIdx.x;
    const int t = threadIdx.x;
    s_out[t] = out[(size_t)n * DIM + t];
    s_h[t]   = h[(size_t)n * DIM + t];
    __syncthreads();
    float d = fmaxf(deg[n], 1.0f);
    float s = agg[(size_t)n * DIM + t] / d + cbias[t];
    #pragma unroll 8
    for (int i = 0; i < DIM; ++i) s += s_out[i] * root[i * DIM + t];
    s_m[t] = fmaxf(s, 0.0f);
    __syncthreads();
    float gi[3], gh[3];
    #pragma unroll
    for (int g = 0; g < 3; ++g) {
        const int j = g * DIM + t;
        float a = b_ih[j], b = b_hh[j];
        const float* wi = w_ih + (size_t)j * DIM;
        const float* wh = w_hh + (size_t)j * DIM;
        #pragma unroll 8
        for (int i = 0; i < DIM; ++i) { a += s_m[i] * wi[i]; b += s_h[i] * wh[i]; }
        gi[g] = a; gh[g] = b;
    }
    float r  = sigm(gi[0] + gh[0]);
    float z  = sigm(gi[1] + gh[1]);
    float nn = tanhf(gi[2] + r * gh[2]);
    float hv = (1.0f - z) * nn + z * s_h[t];
    h[(size_t)n * DIM + t]   = hv;
    out[(size_t)n * DIM + t] = hv;
}

// ---------------- Set2Set LSTM step (per-graph block); also resets ssum/rbuf --------
__global__ __launch_bounds__(64)
void k_lstm(const float* __restrict__ qstar, float* __restrict__ qh, float* __restrict__ qc,
            const float* __restrict__ w_ih, const float* __restrict__ w_hh,
            const float* __restrict__ b_ih, const float* __restrict__ b_hh,
            float* __restrict__ ssum, float* __restrict__ rbuf) {
    __shared__ float s_q[2 * DIM], s_h[DIM];
    const int g = blockIdx.x;
    const int t = threadIdx.x;
    s_h[t] = qh[(size_t)g * DIM + t];
    s_q[t] = qstar[(size_t)g * 2 * DIM + t];
    s_q[DIM + t] = qstar[(size_t)g * 2 * DIM + DIM + t];
    __syncthreads();
    float gate[4];
    #pragma unroll
    for (int k = 0; k < 4; ++k) {
        const int j = k * DIM + t;
        float s = b_ih[j] + b_hh[j];
        const float* wi = w_ih + (size_t)j * (2 * DIM);
        const float* wh = w_hh + (size_t)j * DIM;
        #pragma unroll 8
        for (int i = 0; i < 2 * DIM; ++i) s += s_q[i] * wi[i];
        #pragma unroll 8
        for (int i = 0; i < DIM; ++i) s += s_h[i] * wh[i];
        gate[k] = s;
    }
    float c_old = qc[(size_t)g * DIM + t];
    float c2 = sigm(gate[1]) * c_old + sigm(gate[0]) * tanhf(gate[2]);
    float h2 = sigm(gate[3]) * tanhf(c2);
    qh[(size_t)g * DIM + t] = h2;
    qc[(size_t)g * DIM + t] = c2;
    rbuf[(size_t)g * DIM + t] = 0.0f;
    if (t == 0) ssum[g] = 0.0f;
}

// ---------------- attention logits: enode = exp(out . qh[batch]); ssum += ----------
__global__ void k_attn(const float* __restrict__ out, const float* __restrict__ qh,
                       const int* __restrict__ batch, float* __restrict__ enode,
                       float* __restrict__ ssum) {
    int n = blockIdx.x * blockDim.x + threadIdx.x;
    if (n >= NN) return;
    int g = batch[n];
    const float* orow = out + (size_t)n * DIM;
    const float* hrow = qh + (size_t)g * DIM;
    float s = 0.0f;
    #pragma unroll 8
    for (int i = 0; i < DIM; ++i) s += orow[i] * hrow[i];
    float ex = __expf(s);               // softmax w/o max-shift: mathematically identical
    enode[n] = ex;
    atomicAdd(&ssum[g], ex);
}

// ---------------- weighted readout: rbuf[g] += a * out[n] ----------------
__global__ void k_r(const float* __restrict__ out, const float* __restrict__ enode,
                    const float* __restrict__ ssum, const int* __restrict__ batch,
                    float* __restrict__ rbuf) {
    int idx = blockIdx.x * blockDim.x + threadIdx.x;     // n*64+o
    if (idx >= NN * DIM) return;
    int n = idx >> 6, o = idx & 63;
    int g = batch[n];
    float a = enode[n] / ssum[g];
    atomicAdd(&rbuf[(size_t)g * DIM + o], a * out[idx]);
}

// ---------------- q_star = [qh, r] ----------------
__global__ void k_qstar(const float* __restrict__ qh, const float* __restrict__ rbuf,
                        float* __restrict__ qstar) {
    int idx = blockIdx.x * blockDim.x + threadIdx.x;     // g*128 + j
    if (idx >= NG * 2 * DIM) return;
    int g = idx >> 7, j = idx & 127;
    qstar[idx] = (j < DIM) ? qh[(size_t)g * DIM + j] : rbuf[(size_t)g * DIM + (j - DIM)];
}

// ---------------- final MLP: d_out[g] = lin2(relu(lin1(q_star))) ----------------
__global__ __launch_bounds__(64)
void k_readout(const float* __restrict__ qstar,
               const float* __restrict__ w1, const float* __restrict__ b1,
               const float* __restrict__ w2, const float* __restrict__ b2,
               float* __restrict__ dout) {
    __shared__ float s_o[DIM];
    const int g = blockIdx.x;
    const int t = threadIdx.x;
    const float* q = qstar + (size_t)g * 2 * DIM;
    const float* wr = w1 + (size_t)t * 2 * DIM;
    float s = b1[t];
    #pragma unroll 8
    for (int i = 0; i < 2 * DIM; ++i) s += q[i] * wr[i];
    s_o[t] = fmaxf(s, 0.0f);
    __syncthreads();
    if (t == 0) {
        float acc = b2[0];
        #pragma unroll 8
        for (int i = 0; i < DIM; ++i) acc += s_o[i] * w2[i];
        dout[g] = acc;
    }
}

// ============================== host launcher ==============================
extern "C" void kernel_launch(void* const* d_in, const int* in_sizes, int n_in,
                              void* d_out, int out_size, void* d_ws, size_t ws_size,
                              hipStream_t stream) {
    const float* x         = (const float*)d_in[0];
    const float* edge_attr = (const float*)d_in[1];
    const int*   edge_idx  = (const int*)  d_in[2];   // [2,E] row-major: src=ei[e], dst=ei[E+e]
    const int*   batch     = (const int*)  d_in[3];
    const float* lin0_w    = (const float*)d_in[4];
    const float* lin0_b    = (const float*)d_in[5];
    const float* mlp_w1    = (const float*)d_in[6];
    const float* mlp_b1    = (const float*)d_in[7];
    const float* mlp_w2    = (const float*)d_in[8];
    const float* mlp_b2    = (const float*)d_in[9];
    const float* conv_root = (const float*)d_in[10];
    const float* conv_bias = (const float*)d_in[11];
    const float* gru_w_ih  = (const float*)d_in[12];
    const float* gru_w_hh  = (const float*)d_in[13];
    const float* gru_b_ih  = (const float*)d_in[14];
    const float* gru_b_hh  = (const float*)d_in[15];
    const float* lstm_w_ih = (const float*)d_in[16];
    const float* lstm_w_hh = (const float*)d_in[17];
    const float* lstm_b_ih = (const float*)d_in[18];
    const float* lstm_b_hh = (const float*)d_in[19];
    const float* lin1_w    = (const float*)d_in[20];
    const float* lin1_b    = (const float*)d_in[21];
    const float* lin2_w    = (const float*)d_in[22];
    const float* lin2_b    = (const float*)d_in[23];

    // ---- workspace carve-out (256B aligned) ----
    char* ws = (char*)d_ws;
    size_t off = 0;
    auto carve = [&](size_t bytes) -> char* {
        char* p = ws + off;
        off = (off + bytes + 255) & ~(size_t)255;
        return p;
    };
    unsigned short* ewbf  = (unsigned short*)carve((size_t)EE * EWN * 2);  // 384 MB
    unsigned short* hidbf = (unsigned short*)carve((size_t)EE * HID * 2);  // 12 MB
    unsigned short* w2bf  = (unsigned short*)carve((size_t)EWN * HID * 2); // 1 MB
    float* out_f = (float*)carve((size_t)NN * DIM * 4);
    float* h_f   = (float*)carve((size_t)NN * DIM * 4);
    float* agg   = (float*)carve((size_t)NN * DIM * 4);
    float* deg   = (float*)carve((size_t)NN * 4);
    float* qh    = (float*)carve((size_t)NG * DIM * 4);
    float* qc    = (float*)carve((size_t)NG * DIM * 4);
    float* qstar = (float*)carve((size_t)NG * 2 * DIM * 4);
    float* ssum  = (float*)carve((size_t)NG * 4);
    float* rbuf  = (float*)carve((size_t)NG * DIM * 4);
    float* enode = (float*)carve((size_t)NN * 4);

    const int T = 256;

    // ---- prep: w2 -> bf16; lin0; edge hidden ----
    k_w2bf<<<(EWN * HID + T - 1) / T, T, 0, stream>>>(mlp_w2, w2bf, EWN * HID);
    k_lin0<<<(NN * DIM + T - 1) / T, T, 0, stream>>>(x, lin0_w, lin0_b, out_f, h_f);
    k_edgehid<<<(EE * HID + T - 1) / T, T, 0, stream>>>(edge_attr, mlp_w1, mlp_b1, hidbf);

    // ---- the big GEMM: ew = hid @ w2.T + b2 (WMMA bf16, 2 M-tiles/wave) ----
    {
        dim3 grid(EE / 32, 8);   // 1536 M-pairs x 8 N-groups (32 N-tiles each)
        k_ew_wmma<<<grid, 32, 0, stream>>>(hidbf, w2bf, mlp_b2, ewbf);
    }

    // ---- degree ----
    k_zero<<<(NN + T - 1) / T, T, 0, stream>>>(deg, NN);
    k_deg<<<(EE + T - 1) / T, T, 0, stream>>>(edge_idx, deg);

    // ---- message-passing jumps ----
    for (int j = 0; j < JUMPS; ++j) {
        k_zero<<<(NN * DIM + T - 1) / T, T, 0, stream>>>(agg, NN * DIM);
        k_msg<<<EE, 32, 0, stream>>>(out_f, ewbf, edge_idx, agg);
        k_node<<<NN, 64, 0, stream>>>(agg, deg, conv_root, conv_bias,
                                      gru_w_ih, gru_w_hh, gru_b_ih, gru_b_hh,
                                      out_f, h_f);
    }

    // ---- Set2Set ----
    k_zero<<<(NG * DIM + T - 1) / T, T, 0, stream>>>(qh, NG * DIM);
    k_zero<<<(NG * DIM + T - 1) / T, T, 0, stream>>>(qc, NG * DIM);
    k_zero<<<(NG * 2 * DIM + T - 1) / T, T, 0, stream>>>(qstar, NG * 2 * DIM);
    for (int s = 0; s < S2S; ++s) {
        k_lstm<<<NG, 64, 0, stream>>>(qstar, qh, qc, lstm_w_ih, lstm_w_hh,
                                      lstm_b_ih, lstm_b_hh, ssum, rbuf);
        k_attn<<<(NN + T - 1) / T, T, 0, stream>>>(out_f, qh, batch, enode, ssum);
        k_r<<<(NN * DIM + T - 1) / T, T, 0, stream>>>(out_f, enode, ssum, batch, rbuf);
        k_qstar<<<(NG * 2 * DIM + T - 1) / T, T, 0, stream>>>(qh, rbuf, qstar);
    }

    // ---- readout ----
    k_readout<<<NG, 64, 0, stream>>>(qstar, lin1_w, lin1_b, lin2_w, lin2_b, (float*)d_out);
}
